// ClassicMHA_76656576299170
// MI455X (gfx1250) — matrix operand order repaired and checked by reading, not compiled, not run on hardware
//
#include <hip/hip_runtime.h>
#include <math.h>

#define B_  8
#define N_  4096
#define D_  512
#define H_  8
#define HD_ 64
#define FF_ 2048
#define BN_ (B_ * N_)   // 32768 rows total

typedef __bf16 bf16_t;
typedef __attribute__((ext_vector_type(16))) __bf16 v16bf;
typedef __attribute__((ext_vector_type(8)))  __bf16 v8bf;
typedef __attribute__((ext_vector_type(8)))  float  v8f;

static __device__ __forceinline__ v8f zero8() {
  v8f z = {0.f, 0.f, 0.f, 0.f, 0.f, 0.f, 0.f, 0.f};
  return z;
}

// A-fragment (16x32 bf16, MxK): lane holds row (lane&15); elems 0..7 -> K = h*8+e,
// elems 8..15 -> K = 16+h*8+e  (h = lane>>4). Two 16B loads (global or LDS).
static __device__ __forceinline__ v16bf load_a_frag(const bf16_t* p0, size_t ld, int lane) {
  const bf16_t* p = p0 + (size_t)(lane & 15) * ld + ((lane >> 4) << 3);
  v8bf lo = *(const v8bf*)p;
  v8bf hi = *(const v8bf*)(p + 16);
  v16bf f;
#pragma unroll
  for (int i = 0; i < 8; ++i) { f[i] = lo[i]; f[i + 8] = hi[i]; }
  return f;
}

// B-fragment (32x16 bf16, KxN) read from B^T (N x K row-major): lane holds column
// (lane&15); elems 0..7 -> K = h*8+e, elems 8..15 -> K = 16+h*8+e. Two 16B loads (16B-align safe).
static __device__ __forceinline__ v16bf load_b_frag(const bf16_t* p0, size_t ld, int lane) {
  const bf16_t* p = p0 + (size_t)(lane & 15) * ld + ((lane >> 4) << 3);
  v8bf lo = *(const v8bf*)p;
  v8bf hi = *(const v8bf*)(p + 16);
  v16bf f;
#pragma unroll
  for (int i = 0; i < 8; ++i) { f[i] = lo[i]; f[i + 8] = hi[i]; }
  return f;
}

static __device__ __forceinline__ v8f wmma_bf16(v16bf a, v16bf b, v8f c) {
  return __builtin_amdgcn_wmma_f32_16x16x32_bf16(false, a, false, b, (short)0, c, false, false);
}

// Async global->LDS copy, one 16B chunk per lane. GVS addressing:
// mem = SGPR64(base) + VGPR32(byte offset); LDS dest offset in VDST VGPR.
// Tracked by ASYNCcnt (s_wait_asynccnt).
static __device__ __forceinline__ void async_load_b128(uint32_t lds_off,
                                                       const bf16_t* gbase,
                                                       int32_t gbyteoff) {
  asm volatile("global_load_async_to_lds_b128 %0, %1, %2"
               :
               : "v"(lds_off), "v"(gbyteoff), "s"(gbase)
               : "memory");
}

static __device__ __forceinline__ void wait_async0() {
  asm volatile("s_wait_asynccnt 0x0" ::: "memory");
}

// ---------------------------------------------------------------------------
// Generic GEMM: C[M x Nout] = A[M x K] (bf16, row-major, +acol_off column base)
//               x B (given as B^T [Nout x K] bf16 row-major) + bias.
// Block tile 128x128; 8 waves as 4(M) x 2(N); wave tile 32x64 -> 8 accumulators.
// Double-buffered async global->LDS staging (k-step 32), conflict-free LDS
// padding (row stride 40 bf16 = 20 dwords -> lane bases land on 4-dword grid).
// STORE: 0 = f32 row-major, 1 = bf16 row-major, 2 = bf16 transposed (C^T, ldc=ldT)
// ---------------------------------------------------------------------------
template <int STORE, bool RELU>
__global__ __launch_bounds__(256) void gemm_bf16_kernel(
    const bf16_t* __restrict__ A, int lda_i, int acol_off,
    const bf16_t* __restrict__ BT, int ldb_i,
    const float* __restrict__ bias,
    float* __restrict__ Cf, bf16_t* __restrict__ Cb, int ldc_i, int K) {
  constexpr int LDT = 40;  // padded LDS row stride (elements)
  __shared__ __align__(16) bf16_t As[2][128 * LDT];
  __shared__ __align__(16) bf16_t Bs[2][128 * LDT];

  const size_t lda = (size_t)lda_i, ldb = (size_t)ldb_i, ldc = (size_t)ldc_i;
  const int lane = threadIdx.x & 31;
  const int wave = threadIdx.x >> 5;
  const int wm = wave & 3, wn = wave >> 2;
  const int rowB = blockIdx.y * 128;
  const int colB = blockIdx.x * 128;
  const int t = threadIdx.x;

  // Cooperative tile fill: 128 rows x 64B -> 512 chunks of 16B per tile; each
  // thread moves 2 chunks of A and 2 of B. row = chunk>>2, seg = chunk&3.
  const int cr0 = t >> 2, cs = t & 3;
  const int cr1 = cr0 + 64;
  const uint32_t aB[2] = {(uint32_t)(uintptr_t)&As[0][0], (uint32_t)(uintptr_t)&As[1][0]};
  const uint32_t bB[2] = {(uint32_t)(uintptr_t)&Bs[0][0], (uint32_t)(uintptr_t)&Bs[1][0]};
  const uint32_t ldsA0 = (uint32_t)(cr0 * (LDT * 2) + cs * 16);
  const uint32_t ldsA1 = (uint32_t)(cr1 * (LDT * 2) + cs * 16);

  auto issue = [&](int buf, int k0) {
    const int32_t ga0 = (int32_t)((((size_t)(rowB + cr0) * lda) + acol_off + k0 + cs * 8) * 2);
    const int32_t ga1 = (int32_t)((((size_t)(rowB + cr1) * lda) + acol_off + k0 + cs * 8) * 2);
    async_load_b128(aB[buf] + ldsA0, A, ga0);
    async_load_b128(aB[buf] + ldsA1, A, ga1);
    const int32_t gb0 = (int32_t)((((size_t)(colB + cr0) * ldb) + k0 + cs * 8) * 2);
    const int32_t gb1 = (int32_t)((((size_t)(colB + cr1) * ldb) + k0 + cs * 8) * 2);
    async_load_b128(bB[buf] + ldsA0, BT, gb0);
    async_load_b128(bB[buf] + ldsA1, BT, gb1);
  };

  v8f acc[2][4];
#pragma unroll
  for (int i = 0; i < 2; ++i)
#pragma unroll
    for (int j = 0; j < 4; ++j) acc[i][j] = zero8();

  issue(0, 0);
  int buf = 0;
  for (int k0 = 0; k0 < K; k0 += 32) {
    wait_async0();        // this wave's async chunks have landed in LDS
    __syncthreads();      // every wave's chunks have landed -> tile complete
    if (k0 + 32 < K) issue(buf ^ 1, k0 + 32);  // prefetch next k-tile

    const bf16_t* Ab = &As[buf][(wm * 32) * LDT];
    const bf16_t* Bb = &Bs[buf][(wn * 64) * LDT];
    v16bf a0 = load_a_frag(Ab, LDT, lane);
    v16bf a1 = load_a_frag(Ab + 16 * LDT, LDT, lane);
#pragma unroll
    for (int j = 0; j < 4; ++j) {
      v16bf bfr = load_b_frag(Bb + (size_t)(j * 16) * LDT, LDT, lane);
      acc[0][j] = wmma_bf16(a0, bfr, acc[0][j]);
      acc[1][j] = wmma_bf16(a1, bfr, acc[1][j]);
    }
    __syncthreads();      // all waves done reading this buffer
    buf ^= 1;
  }

  const int row0 = rowB + wm * 32;
  const int col0c = colB + wn * 64;
#pragma unroll
  for (int mi = 0; mi < 2; ++mi) {
#pragma unroll
    for (int j = 0; j < 4; ++j) {
      const int col = col0c + j * 16 + (lane & 15);
      const float bv = bias ? bias[col] : 0.0f;
      const int rbase = row0 + mi * 16 + ((lane >> 4) << 3);
      if (STORE == 2) {
        // C fragment: lane = fixed column, 8 consecutive rows -> one packed store
        v8bf pk;
#pragma unroll
        for (int r = 0; r < 8; ++r) {
          float v = acc[mi][j][r] + bv;
          if (RELU) v = fmaxf(v, 0.f);
          pk[r] = (bf16_t)v;
        }
        *(v8bf*)(Cb + (size_t)col * ldc + rbase) = pk;
      } else {
#pragma unroll
        for (int r = 0; r < 8; ++r) {
          float v = acc[mi][j][r] + bv;
          if (RELU) v = fmaxf(v, 0.f);
          if (STORE == 0) Cf[(size_t)(rbase + r) * ldc + col] = v;
          else            Cb[(size_t)(rbase + r) * ldc + col] = (bf16_t)v;
        }
      }
    }
  }
}

// ---------------------------------------------------------------------------
// Per-(b,h): dot = K^T V / sqrt(HD) over N=4096, softmax rows, write attn^T bf16.
// KT/VT laid out (512 x BN): row = h*64+d, col = b*4096+n, so both WMMA operands
// load with contiguous-K fragments. 8 waves split N; reduce via ds_add_f32.
// ---------------------------------------------------------------------------
__global__ __launch_bounds__(256) void ktv_softmax_kernel(
    const bf16_t* __restrict__ KTm, const bf16_t* __restrict__ VTm,
    bf16_t* __restrict__ attnT) {
  __shared__ float sdot[HD_ * HD_];  // 16 KB
  const int bh = blockIdx.x;
  const int b = bh >> 3, h = bh & 7;
  const int lane = threadIdx.x & 31, wave = threadIdx.x >> 5;

  for (int i = threadIdx.x; i < HD_ * HD_; i += 256) sdot[i] = 0.f;
  __syncthreads();

  const bf16_t* Kb = KTm + (size_t)(h * HD_) * BN_ + (size_t)b * N_ + wave * 512;
  const bf16_t* Vb = VTm + (size_t)(h * HD_) * BN_ + (size_t)b * N_ + wave * 512;

  v8f acc[4][4];
#pragma unroll
  for (int i = 0; i < 4; ++i)
#pragma unroll
    for (int j = 0; j < 4; ++j) acc[i][j] = zero8();

  for (int k0 = 0; k0 < 512; k0 += 32) {
    v16bf a[4], bb[4];
#pragma unroll
    for (int i = 0; i < 4; ++i) a[i] = load_a_frag(Kb + (size_t)(i * 16) * BN_ + k0, BN_, lane);
#pragma unroll
    for (int j = 0; j < 4; ++j) bb[j] = load_b_frag(Vb + (size_t)(j * 16) * BN_ + k0, BN_, lane);
#pragma unroll
    for (int i = 0; i < 4; ++i)
#pragma unroll
      for (int j = 0; j < 4; ++j) acc[i][j] = wmma_bf16(a[i], bb[j], acc[i][j]);
  }

#pragma unroll
  for (int i = 0; i < 4; ++i) {
#pragma unroll
    for (int j = 0; j < 4; ++j) {
      const int col = j * 16 + (lane & 15);
      const int rbase = i * 16 + ((lane >> 4) << 3);
#pragma unroll
      for (int r = 0; r < 8; ++r)
        atomicAdd(&sdot[(rbase + r) * HD_ + col], acc[i][j][r]);
    }
  }
  __syncthreads();

  if (threadIdx.x < HD_) {
    const int dd = threadIdx.x;
    const float scale = 0.125f;  // 1/sqrt(64)
    float mx = -3.4e38f;
    for (int e = 0; e < HD_; ++e) mx = fmaxf(mx, sdot[dd * HD_ + e] * scale);
    float s = 0.f;
    for (int e = 0; e < HD_; ++e) {
      float v = expf(sdot[dd * HD_ + e] * scale - mx);
      sdot[dd * HD_ + e] = v;
      s += v;
    }
    const float inv = 1.f / s;
    bf16_t* at = attnT + (size_t)bh * HD_ * HD_;
    for (int e = 0; e < HD_; ++e)
      at[(size_t)e * HD_ + dd] = (bf16_t)(sdot[dd * HD_ + e] * inv);  // store transposed
  }
}

// ---------------------------------------------------------------------------
// y1[b,n,h*64+e] = sum_d Q[b,n,h*64+d] * attn[b,h,d,e]   (K = 64, 2 WMMA steps)
// ---------------------------------------------------------------------------
__global__ __launch_bounds__(256) void qattn_kernel(
    const bf16_t* __restrict__ Qm, const bf16_t* __restrict__ attnT,
    bf16_t* __restrict__ y1) {
  const int h = blockIdx.y;
  const int lane = threadIdx.x & 31, wave = threadIdx.x >> 5;
  const int row0 = blockIdx.x * 256 + wave * 32;
  const int b = row0 >> 12;  // 4096 rows per batch; tiles never straddle batches
  const bf16_t* At = attnT + (size_t)((b * H_ + h) * HD_) * HD_;
  const bf16_t* Ab = Qm + (size_t)row0 * D_ + h * HD_;

  v8f acc[2][4];
#pragma unroll
  for (int i = 0; i < 2; ++i)
#pragma unroll
    for (int j = 0; j < 4; ++j) acc[i][j] = zero8();

#pragma unroll
  for (int k0 = 0; k0 < HD_; k0 += 32) {
    v16bf a0 = load_a_frag(Ab + k0, D_, lane);
    v16bf a1 = load_a_frag(Ab + (size_t)16 * D_ + k0, D_, lane);
#pragma unroll
    for (int j = 0; j < 4; ++j) {
      v16bf bfr = load_b_frag(At + (size_t)(j * 16) * HD_ + k0, HD_, lane);
      acc[0][j] = wmma_bf16(a0, bfr, acc[0][j]);
      acc[1][j] = wmma_bf16(a1, bfr, acc[1][j]);
    }
  }

#pragma unroll
  for (int mi = 0; mi < 2; ++mi) {
#pragma unroll
    for (int j = 0; j < 4; ++j) {
      const int col = h * HD_ + j * 16 + (lane & 15);
      const int rbase = row0 + mi * 16 + ((lane >> 4) << 3);
#pragma unroll
      for (int r = 0; r < 8; ++r)
        y1[(size_t)(rbase + r) * D_ + col] = (bf16_t)acc[mi][j][r];
    }
  }
}

// ---------------------------------------------------------------------------
// LayerNorm over D=512 cols, one 256-thread block per row, 2 elems/thread.
// MODE 0: t = 2*X[row] ; MODE 1: t = X[row] + R[row].
// ---------------------------------------------------------------------------
template <int MODE>
__global__ __launch_bounds__(256) void layernorm_kernel(
    const float* __restrict__ X, const float* __restrict__ R,
    const float* __restrict__ g, const float* __restrict__ be,
    float* __restrict__ outF, bf16_t* __restrict__ outB) {
  const int row = blockIdx.x;
  const int t = threadIdx.x;
  const float* x = X + (size_t)row * D_;
  float v0, v1;
  if (MODE == 0) {
    v0 = 2.f * x[t];
    v1 = 2.f * x[t + 256];
  } else {
    const float* rr = R + (size_t)row * D_;
    v0 = x[t] + rr[t];
    v1 = x[t + 256] + rr[t + 256];
  }
  __shared__ float red[256];
  red[t] = v0 + v1;
  __syncthreads();
  for (int s = 128; s > 0; s >>= 1) {
    if (t < s) red[t] += red[t + s];
    __syncthreads();
  }
  const float mean = red[0] * (1.0f / D_);
  __syncthreads();
  const float d0 = v0 - mean, d1 = v1 - mean;
  red[t] = d0 * d0 + d1 * d1;
  __syncthreads();
  for (int s = 128; s > 0; s >>= 1) {
    if (t < s) red[t] += red[t + s];
    __syncthreads();
  }
  const float rstd = rsqrtf(red[0] * (1.0f / D_) + 1e-5f);
  const float o0 = d0 * rstd * g[t] + be[t];
  const float o1 = d1 * rstd * g[t + 256] + be[t + 256];
  outF[(size_t)row * D_ + t] = o0;
  outF[(size_t)row * D_ + t + 256] = o1;
  if (outB) {
    outB[(size_t)row * D_ + t] = (bf16_t)o0;
    outB[(size_t)row * D_ + t + 256] = (bf16_t)o1;
  }
}

// ---------------------------------------------------------------------------
// Tiled transpose f32(R x C) -> bf16(C x R), batched along blockIdx.z.
// ---------------------------------------------------------------------------
__global__ __launch_bounds__(256) void transpose_f32_to_bf16_kernel(
    const float* __restrict__ src, bf16_t* __restrict__ dst, int R, int C) {
  __shared__ float tile[32][33];
  const float* s = src + (size_t)blockIdx.z * R * C;
  bf16_t* d = dst + (size_t)blockIdx.z * R * C;
  const int c0 = blockIdx.x * 32, r0 = blockIdx.y * 32;
  const int tx = threadIdx.x, ty = threadIdx.y;  // block (32,8)
#pragma unroll
  for (int i = 0; i < 4; ++i)
    tile[ty + i * 8][tx] = s[(size_t)(r0 + ty + i * 8) * C + c0 + tx];
  __syncthreads();
#pragma unroll
  for (int i = 0; i < 4; ++i)
    d[(size_t)(c0 + ty + i * 8) * R + r0 + tx] = (bf16_t)tile[tx][ty + i * 8];
}

// res (B*N x D) f32 -> out (B x D x N) f32, tiled transpose per batch.
__global__ __launch_bounds__(256) void transpose_out_kernel(
    const float* __restrict__ res, float* __restrict__ out) {
  __shared__ float tile[32][33];
  const int b = blockIdx.z;
  const int d0 = blockIdx.x * 32, n0 = blockIdx.y * 32;
  const int tx = threadIdx.x, ty = threadIdx.y;
  const float* s = res + (size_t)b * N_ * D_;
  float* o = out + (size_t)b * D_ * N_;
#pragma unroll
  for (int i = 0; i < 4; ++i)
    tile[ty + i * 8][tx] = s[(size_t)(n0 + ty + i * 8) * D_ + d0 + tx];
  __syncthreads();
#pragma unroll
  for (int i = 0; i < 4; ++i)
    o[(size_t)(d0 + ty + i * 8) * N_ + n0 + tx] = tile[tx][ty + i * 8];
}

// ---------------------------------------------------------------------------
extern "C" void kernel_launch(void* const* d_in, const int* in_sizes, int n_in,
                              void* d_out, int out_size, void* d_ws, size_t ws_size,
                              hipStream_t stream) {
  (void)in_sizes; (void)n_in; (void)out_size; (void)ws_size;
  const float* x   = (const float*)d_in[0];
  const float* Wq  = (const float*)d_in[1];  const float* bq  = (const float*)d_in[2];
  const float* Wk  = (const float*)d_in[3];  const float* bk  = (const float*)d_in[4];
  const float* Wv  = (const float*)d_in[5];  const float* bv  = (const float*)d_in[6];
  const float* Wo  = (const float*)d_in[7];  const float* bo  = (const float*)d_in[8];
  const float* W1  = (const float*)d_in[9];  const float* b1  = (const float*)d_in[10];
  const float* W2  = (const float*)d_in[11]; const float* b2  = (const float*)d_in[12];
  const float* g1  = (const float*)d_in[13]; const float* be1 = (const float*)d_in[14];
  const float* g2  = (const float*)d_in[15]; const float* be2 = (const float*)d_in[16];

  char* ws = (char*)d_ws;
  size_t off = 0;
  auto alloc = [&](size_t bytes) -> void* {
    void* p = (void*)(ws + off);
    off += (bytes + 255) & ~(size_t)255;
    return p;
  };

  bf16_t* xt    = (bf16_t*)alloc((size_t)BN_ * D_ * 2);   // x transposed, bf16
  bf16_t* WqT   = (bf16_t*)alloc((size_t)D_ * D_ * 2);
  bf16_t* WkT   = (bf16_t*)alloc((size_t)D_ * D_ * 2);
  bf16_t* WvT   = (bf16_t*)alloc((size_t)D_ * D_ * 2);
  bf16_t* WoT   = (bf16_t*)alloc((size_t)D_ * D_ * 2);
  bf16_t* W1T   = (bf16_t*)alloc((size_t)FF_ * D_ * 2);
  bf16_t* W2T   = (bf16_t*)alloc((size_t)D_ * FF_ * 2);
  bf16_t* Qm    = (bf16_t*)alloc((size_t)BN_ * D_ * 2);
  bf16_t* KTm   = (bf16_t*)alloc((size_t)D_ * BN_ * 2);   // K^T per channel
  bf16_t* VTm   = (bf16_t*)alloc((size_t)D_ * BN_ * 2);   // V^T per channel
  bf16_t* attnT = (bf16_t*)alloc((size_t)B_ * H_ * HD_ * HD_ * 2);
  bf16_t* y1    = (bf16_t*)alloc((size_t)BN_ * D_ * 2);
  float*  y2    = (float*)alloc((size_t)BN_ * D_ * 4);    // reused as y (post-LN1) f32
  bf16_t* ybf   = (bf16_t*)alloc((size_t)BN_ * D_ * 2);
  bf16_t* hdd   = (bf16_t*)alloc((size_t)BN_ * FF_ * 2);
  float*  zf    = (float*)alloc((size_t)BN_ * D_ * 4);    // reused as final residual f32

  const dim3 tb(32, 8);
  // Prep: transpose+convert activations and weights to bf16 (weights -> W^T).
  transpose_f32_to_bf16_kernel<<<dim3(N_ / 32, D_ / 32, B_), tb, 0, stream>>>(x, xt, D_, N_);
  transpose_f32_to_bf16_kernel<<<dim3(D_ / 32, D_ / 32, 1), tb, 0, stream>>>(Wq, WqT, D_, D_);
  transpose_f32_to_bf16_kernel<<<dim3(D_ / 32, D_ / 32, 1), tb, 0, stream>>>(Wk, WkT, D_, D_);
  transpose_f32_to_bf16_kernel<<<dim3(D_ / 32, D_ / 32, 1), tb, 0, stream>>>(Wv, WvT, D_, D_);
  transpose_f32_to_bf16_kernel<<<dim3(D_ / 32, D_ / 32, 1), tb, 0, stream>>>(Wo, WoT, D_, D_);
  transpose_f32_to_bf16_kernel<<<dim3(FF_ / 32, D_ / 32, 1), tb, 0, stream>>>(W1, W1T, D_, FF_);
  transpose_f32_to_bf16_kernel<<<dim3(D_ / 32, FF_ / 32, 1), tb, 0, stream>>>(W2, W2T, FF_, D_);

  const dim3 g512(D_ / 128, BN_ / 128);  // (4, 256)
  // Q normal; K,V stored transposed via C^T store mode (coalesced packed stores).
  gemm_bf16_kernel<1, false><<<g512, 256, 0, stream>>>(xt, D_, 0, WqT, D_, bq, nullptr, Qm, D_, D_);
  gemm_bf16_kernel<2, false><<<g512, 256, 0, stream>>>(xt, D_, 0, WkT, D_, bk, nullptr, KTm, BN_, D_);
  gemm_bf16_kernel<2, false><<<g512, 256, 0, stream>>>(xt, D_, 0, WvT, D_, bv, nullptr, VTm, BN_, D_);

  ktv_softmax_kernel<<<B_ * H_, 256, 0, stream>>>(KTm, VTm, attnT);
  qattn_kernel<<<dim3(BN_ / 256, H_), 256, 0, stream>>>(Qm, attnT, y1);

  gemm_bf16_kernel<0, false><<<g512, 256, 0, stream>>>(y1, D_, 0, WoT, D_, bo, y2, nullptr, D_, D_);
  layernorm_kernel<0><<<BN_, 256, 0, stream>>>(y2, nullptr, g1, be1, y2, ybf);
  gemm_bf16_kernel<1, true><<<dim3(FF_ / 128, BN_ / 128), 256, 0, stream>>>(ybf, D_, 0, W1T, D_, b1,
                                                                            nullptr, hdd, FF_, D_);
  gemm_bf16_kernel<0, false><<<g512, 256, 0, stream>>>(hdd, FF_, 0, W2T, FF_, b2, zf, nullptr, D_, FF_);
  layernorm_kernel<1><<<BN_, 256, 0, stream>>>(y2, zf, g2, be2, zf, nullptr);
  transpose_out_kernel<<<dim3(D_ / 32, N_ / 32, B_), tb, 0, stream>>>(zf, (float*)d_out);
}
